// MultiQueryAttention_66571993088381
// MI455X (gfx1250) — matrix operand order repaired and checked by reading, not compiled, not run on hardware
//
#include <hip/hip_runtime.h>
#include <hip/hip_bf16.h>

typedef _Float16 half_t;
typedef __attribute__((ext_vector_type(16))) _Float16 v16h;
typedef __attribute__((ext_vector_type(8)))  _Float16 v8h;
typedef __attribute__((ext_vector_type(8)))  float    v8f;

#define D_MODEL   2048
#define HEAD_DIM  128
#define NUM_HEADS 16
#define SEQ_L     2048
#define BATCH     2

// ---------------------------------------------------------------------------
// CDNA5 async global->LDS copy (ASYNCcnt path), with sync fallback
// ---------------------------------------------------------------------------
#if defined(__has_builtin)
#if __has_builtin(__builtin_amdgcn_global_load_async_to_lds_b128)
#define HAVE_ASYNC_LDS 1
#endif
#endif

typedef int v4i __attribute__((__vector_size__(16)));
typedef __attribute__((address_space(1))) void glob_void;
typedef __attribute__((address_space(3))) void lds_void;
typedef __attribute__((address_space(1))) v4i* gvec_p;
typedef __attribute__((address_space(3))) v4i* lvec_p;

// copy 16 bytes global -> LDS
static __device__ inline void cp16_g2l(const half_t* g, half_t* l) {
#if defined(HAVE_ASYNC_LDS)
  __builtin_amdgcn_global_load_async_to_lds_b128(
      (gvec_p)(glob_void*)g, (lvec_p)(lds_void*)l, /*offset=*/0, /*cpol=*/0);
#else
  v8h v = *(const v8h*)g;
  *(v8h*)l = v;
#endif
}

#if defined(HAVE_ASYNC_LDS)
// wave issued 8 copies per stage: keep newest 8 outstanding -> older stage done
#define WAIT_ASYNC_KEEP8() asm volatile("s_wait_asynccnt 0x8" ::: "memory")
#define WAIT_ASYNC_ALL()   asm volatile("s_wait_asynccnt 0x0" ::: "memory")
#else
#define WAIT_ASYNC_KEEP8()
#define WAIT_ASYNC_ALL()
#endif

// ---------------------------------------------------------------------------
// WMMA helpers (CDNA5 v_wmma_f32_16x16x32_f16, wave32)
// ---------------------------------------------------------------------------
static __device__ inline v8f wmma_f16(v16h a, v16h b, v8f c) {
  return __builtin_amdgcn_wmma_f32_16x16x32_f16(false, a, false, b, (short)0, c,
                                                false, false);
}

// A-matrix 16x32 f16 fragment (contiguous K order, matching B):
// lanes 0-15 : row M=lane,    halves 0..15 = K k0..k0+15
// lanes 16-31: row M=lane-16, halves 0..15 = K k0+16..k0+31
static __device__ inline v16h load_a_frag(const half_t* base, int ld, int k0, int lane) {
  int r  = lane & 15;
  int ko = k0 + ((lane & 16) ? 16 : 0);
  const half_t* p = base + (size_t)r * ld + ko;
  union { v16h v; v8h h[2]; } u;
  u.h[0] = *(const v8h*)(p);
  u.h[1] = *(const v8h*)(p + 8);
  return u.v;
}

// B-matrix 32x16 f16 fragment where B[k][n] lives at rowsrc[n*ld + k]:
// lanes 0-15 : col N=lane,    halves = K k0..k0+15
// lanes 16-31: col N=lane-16, halves = K k0+16..k0+31
static __device__ inline v16h load_b_frag(const half_t* rowsrc, int ld, int k0, int lane) {
  int n  = lane & 15;
  int ko = k0 + ((lane & 16) ? 16 : 0);
  return *(const v16h*)(rowsrc + (size_t)n * ld + ko);
}

// ---------------------------------------------------------------------------
// f32 -> f16 conversion (vectorized x4)
// ---------------------------------------------------------------------------
struct __align__(8) H4 { half_t x, y, z, w; };

__global__ __launch_bounds__(256) void cvt_f32_f16(const float* __restrict__ in,
                                                   half_t* __restrict__ out, int n4) {
  int i = blockIdx.x * blockDim.x + threadIdx.x;
  if (i < n4) {
    float4 v = ((const float4*)in)[i];
    H4 h = { (half_t)v.x, (half_t)v.y, (half_t)v.z, (half_t)v.w };
    ((H4*)out)[i] = h;
  }
}

// ---------------------------------------------------------------------------
// GEMM: Y[m][n] = sum_k X[m][k]*W[n][k] + bias[n]   (torch Linear)
// Each wave: 32x64 C block = 2 A-frags x 4 B-frags x 8 WMMAs per 32-wide K step.
// transpose_out==1 writes Y as Vt[b][n][l] (n-major) for the attention P*V GEMM.
// ---------------------------------------------------------------------------
__global__ __launch_bounds__(128) void gemm_xwT(const half_t* __restrict__ X,
                                                const half_t* __restrict__ W,
                                                const float* __restrict__ bias,
                                                half_t* __restrict__ Y,
                                                int M, int N, int transpose_out) {
  const int K = D_MODEL;
  int lane = threadIdx.x & 31;
  int wave = blockIdx.x * (blockDim.x >> 5) + (threadIdx.x >> 5);
  int mtiles = M >> 5;
  int m0 = (wave % mtiles) << 5;
  int n0 = (wave / mtiles) << 6;

  v8f acc[2][4];
#pragma unroll
  for (int nt = 0; nt < 4; ++nt) {
    float bv = bias[n0 + nt * 16 + (lane & 15)];
#pragma unroll
    for (int mt = 0; mt < 2; ++mt)
#pragma unroll
      for (int j = 0; j < 8; ++j) acc[mt][nt][j] = bv;
  }

  const half_t* Xa0 = X + (size_t)m0 * K;
  const half_t* Xa1 = X + (size_t)(m0 + 16) * K;
  for (int k0 = 0; k0 < K; k0 += 32) {
    v16h a0 = load_a_frag(Xa0, K, k0, lane);
    v16h a1 = load_a_frag(Xa1, K, k0, lane);
#pragma unroll
    for (int nt = 0; nt < 4; ++nt) {
      v16h b = load_b_frag(W + (size_t)(n0 + nt * 16) * K, K, k0, lane);
      acc[0][nt] = wmma_f16(a0, b, acc[0][nt]);
      acc[1][nt] = wmma_f16(a1, b, acc[1][nt]);
    }
  }

  int nlo  = lane & 15;
  int rsel = (lane & 16) ? 8 : 0;
#pragma unroll
  for (int mt = 0; mt < 2; ++mt)
#pragma unroll
    for (int nt = 0; nt < 4; ++nt)
#pragma unroll
      for (int j = 0; j < 8; ++j) {
        int m = m0 + mt * 16 + j + rsel;
        int n = n0 + nt * 16 + nlo;
        size_t idx;
        if (transpose_out) {                       // Vt[b][n][l]
          int bb = m >> 11;                        // m / SEQ_L
          int l  = m & (SEQ_L - 1);
          idx = ((size_t)bb * N + n) * SEQ_L + l;
        } else {
          idx = (size_t)m * N + n;
        }
        Y[idx] = (half_t)acc[mt][nt][j];
      }
}

// ---------------------------------------------------------------------------
// Flash-style MQA attention. 4 waves/block on adjacent q-tiles of one (b,h);
// the block shares K/V: 32-key blocks are double-buffered in LDS via
// async global->LDS copies overlapped with WMMA compute.
// Qh: [B*L, 2048] f16 (== [b,l,h,d]); Kh: [B*L, 128] f16; Vt: [B,128,L] f16.
// ---------------------------------------------------------------------------
__global__ __launch_bounds__(128) void mqa_attn(const half_t* __restrict__ Qh,
                                                const half_t* __restrict__ Kh,
                                                const half_t* __restrict__ Vt,
                                                float* __restrict__ Out) {
  __shared__ __align__(32) half_t Ks[2][32 * HEAD_DIM];   // 2 x 8KB  [key][d]
  __shared__ __align__(32) half_t Vs[2][HEAD_DIM * 32];   // 2 x 8KB  [d][key]
  __shared__ __align__(32) half_t pbuf[4][16 * 32];       // per-wave P staging

  int lane  = threadIdx.x & 31;
  int wslot = threadIdx.x >> 5;
  int wave  = blockIdx.x * 4 + wslot;

  const int qtiles = SEQ_L / 16;                   // 128
  int q0 = (wave % qtiles) * 16;
  int h  = (wave / qtiles) % NUM_HEADS;
  int b  = wave / (qtiles * NUM_HEADS);

  const float scale = 0.08838834764831845f;        // 1/sqrt(128)

  const half_t* Kb = Kh + (size_t)b * SEQ_L * HEAD_DIM;
  const half_t* Vb = Vt + (size_t)b * HEAD_DIM * SEQ_L;

  // cooperative stage of one 32-key block into LDS buffer p (8x16B per thread)
  auto stage = [&](int kb, int p) {
    int t = threadIdx.x;                           // 0..127
    const half_t* ksrc = Kb + (size_t)kb * 32 * HEAD_DIM;  // contiguous 8KB
    const half_t* vsrc = Vb + kb * 32;
#pragma unroll
    for (int i = 0; i < 4; ++i) {                  // K block, linear copy
      int ci = t + i * 128;                        // 16B chunk id, 0..511
      cp16_g2l(ksrc + ci * 8, &Ks[p][ci * 8]);
    }
#pragma unroll
    for (int i = 0; i < 4; ++i) {                  // V block, 128 rows x 64B
      int ci = t + i * 128;
      int d = ci >> 2, off = (ci & 3) * 8;
      cp16_g2l(vsrc + (size_t)d * SEQ_L + off, &Vs[p][ci * 8]);
    }
  };

  // Q tile 16x128 held in registers as 4 A-fragments
  const half_t* Qbase = Qh + ((size_t)(b * SEQ_L + q0)) * D_MODEL + h * HEAD_DIM;
  v16h aQ[4];
#pragma unroll
  for (int d = 0; d < 4; ++d) aQ[d] = load_a_frag(Qbase, D_MODEL, d * 32, lane);

  v8f zero8 = {};
  v8f o[8];
#pragma unroll
  for (int dt = 0; dt < 8; ++dt) o[dt] = zero8;
  float mrow[8], lrow[8];
#pragma unroll
  for (int j = 0; j < 8; ++j) { mrow[j] = -1e30f; lrow[j] = 0.f; }

  half_t* pb = pbuf[wslot];
  int col0 = lane & 15;
  int rsel = (lane & 16) ? 8 : 0;

  stage(0, 0);                                     // prefetch first block

  for (int kb = 0; kb < SEQ_L / 32; ++kb) {
    int p = kb & 1;
    if (kb + 1 < SEQ_L / 32) {
      stage(kb + 1, p ^ 1);                        // prefetch next block
      WAIT_ASYNC_KEEP8();                          // current block's copies done
    } else {
      WAIT_ASYNC_ALL();
    }
    __syncthreads();                               // all waves' copies visible

    // ---- S = Q * K^T for 32 key columns (two 16-wide C tiles) ----
    v8f s0 = zero8, s1 = zero8;
#pragma unroll
    for (int d = 0; d < 4; ++d) {
      const half_t* kp = &Ks[p][(size_t)(lane & 15) * HEAD_DIM
                                + d * 32 + ((lane & 16) ? 16 : 0)];
      v16h b0 = *(const v16h*)kp;
      v16h b1 = *(const v16h*)(kp + 16 * HEAD_DIM);
      s0 = wmma_f16(aQ[d], b0, s0);
      s1 = wmma_f16(aQ[d], b1, s1);
    }

    // ---- online softmax over the 32 new columns ----
#pragma unroll
    for (int j = 0; j < 8; ++j) {
      float v0 = s0[j] * scale;
      float v1 = s1[j] * scale;
      float rm = fmaxf(v0, v1);
#pragma unroll
      for (int off = 8; off >= 1; off >>= 1)
        rm = fmaxf(rm, __shfl_xor(rm, off, 32));   // reduce within 16-lane half
      float mnew = fmaxf(mrow[j], rm);
      float corr = __expf(mrow[j] - mnew);
      float p0 = __expf(v0 - mnew);
      float p1 = __expf(v1 - mnew);
      float rs = p0 + p1;
#pragma unroll
      for (int off = 8; off >= 1; off >>= 1)
        rs += __shfl_xor(rs, off, 32);
      lrow[j] = lrow[j] * corr + rs;
      mrow[j] = mnew;
#pragma unroll
      for (int dt = 0; dt < 8; ++dt) o[dt][j] *= corr;
      int r = j + rsel;                            // C-layout row for this half
      pb[r * 32 + col0]      = (half_t)p0;
      pb[r * 32 + 16 + col0] = (half_t)p1;
    }

    // LDS RAW hazard: stores above feed cross-lane A-fragment loads below
    asm volatile("s_wait_dscnt 0" ::: "memory");

    // ---- O += P * V over this 32-key block ----
    v16h aP = load_a_frag(pb, 32, 0, lane);
#pragma unroll
    for (int dt = 0; dt < 8; ++dt) {
      const half_t* vp = &Vs[p][(size_t)(dt * 16 + (lane & 15)) * 32
                                + ((lane & 16) ? 16 : 0)];
      v16h bv = *(const v16h*)vp;
      o[dt] = wmma_f16(aP, bv, o[dt]);
    }

    __syncthreads();                               // block done reading buffer p
  }

  // ---- epilogue: normalize and store f32 ----
  float inv[8];
#pragma unroll
  for (int j = 0; j < 8; ++j) inv[j] = 1.f / lrow[j];
  float* ob = Out + ((size_t)(b * SEQ_L + q0)) * D_MODEL + h * HEAD_DIM;
#pragma unroll
  for (int dt = 0; dt < 8; ++dt)
#pragma unroll
    for (int j = 0; j < 8; ++j)
      ob[(size_t)(j + rsel) * D_MODEL + dt * 16 + col0] = o[dt][j] * inv[j];
}

// ---------------------------------------------------------------------------
extern "C" void kernel_launch(void* const* d_in, const int* in_sizes, int n_in,
                              void* d_out, int out_size, void* d_ws, size_t ws_size,
                              hipStream_t stream) {
  (void)in_sizes; (void)n_in; (void)out_size; (void)ws_size;
  const float* x    = (const float*)d_in[0];
  const float* Wq_w = (const float*)d_in[1];
  const float* Wq_b = (const float*)d_in[2];
  const float* Wk_w = (const float*)d_in[3];
  const float* Wk_b = (const float*)d_in[4];
  const float* Wv_w = (const float*)d_in[5];
  const float* Wv_b = (const float*)d_in[6];
  float* out = (float*)d_out;

  const size_t M = (size_t)BATCH * SEQ_L;          // 4096

  // workspace layout (halves): total ~21.5M halves = 43 MB
  half_t* xh  = (half_t*)d_ws;                     // [M, 2048]
  half_t* wqh = xh  + M * D_MODEL;                 // [2048, 2048]
  half_t* wkh = wqh + (size_t)D_MODEL * D_MODEL;   // [128, 2048]
  half_t* wvh = wkh + (size_t)HEAD_DIM * D_MODEL;  // [128, 2048]
  half_t* qh  = wvh + (size_t)HEAD_DIM * D_MODEL;  // [M, 2048]
  half_t* kh  = qh  + M * D_MODEL;                 // [M, 128]
  half_t* vth = kh  + M * HEAD_DIM;                // [B, 128, 2048]

  // 1) convert inputs to f16
  {
    int n4 = (int)(M * D_MODEL / 4);
    cvt_f32_f16<<<(n4 + 255) / 256, 256, 0, stream>>>(x, xh, n4);
  }
  {
    int n4 = D_MODEL * D_MODEL / 4;
    cvt_f32_f16<<<(n4 + 255) / 256, 256, 0, stream>>>(Wq_w, wqh, n4);
  }
  {
    int n4 = HEAD_DIM * D_MODEL / 4;
    cvt_f32_f16<<<(n4 + 255) / 256, 256, 0, stream>>>(Wk_w, wkh, n4);
    cvt_f32_f16<<<(n4 + 255) / 256, 256, 0, stream>>>(Wv_w, wvh, n4);
  }

  // 2) projections (WMMA GEMMs), 4 waves / block
  {
    int waves = (int)(M / 32) * (D_MODEL / 64);    // 4096
    gemm_xwT<<<waves / 4, 128, 0, stream>>>(xh, wqh, Wq_b, qh, (int)M, D_MODEL, 0);
  }
  {
    int waves = (int)(M / 32) * (HEAD_DIM / 64);   // 256
    gemm_xwT<<<waves / 4, 128, 0, stream>>>(xh, wkh, Wk_b, kh,  (int)M, HEAD_DIM, 0);
    gemm_xwT<<<waves / 4, 128, 0, stream>>>(xh, wvh, Wv_b, vth, (int)M, HEAD_DIM, 1);
  }

  // 3) flash attention (WMMA), one wave per (b, h, q-tile)
  {
    int waves = BATCH * NUM_HEADS * (SEQ_L / 16);  // 4096
    mqa_attn<<<waves / 4, 128, 0, stream>>>(qh, kh, vth, out);
  }
}
